// TimeSformerEncoder_46162308497916
// MI455X (gfx1250) — matrix operand order repaired and checked by reading, not compiled, not run on hardware
//
#include <hip/hip_runtime.h>

// ---------------------------------------------------------------------------
// TimeSformer forward for MI455X (gfx1250), wave32 + WMMA bf16 + async-to-LDS.
// All matmuls run through v_wmma_f32_16x16x32_bf16 (fp32 accumulate).
// GEMM tiles and space-attention K/V tiles are staged into LDS with
// GLOBAL_LOAD_ASYNC_TO_LDS_B128, synchronized with s_wait_asynccnt.
// ---------------------------------------------------------------------------

typedef __bf16 bhalf;
typedef bhalf v16bf __attribute__((ext_vector_type(16)));
typedef float v8f   __attribute__((ext_vector_type(8)));
typedef int   v4i_t __attribute__((vector_size(16)));   // matches builtin's int4

#define ASG __attribute__((address_space(1)))
#define ASL __attribute__((address_space(3)))

#if defined(__gfx1250__) && __has_builtin(__builtin_amdgcn_global_load_async_to_lds_b128) && __has_builtin(__builtin_amdgcn_s_wait_asynccnt)
#define TSF_ASYNC 1
#else
#define TSF_ASYNC 0
#endif

union FragB16 {
    v16bf v;
    unsigned int   w[8];
    unsigned short u[16];
};

__device__ __forceinline__ unsigned short f2bf(float f) {
    unsigned int x = __builtin_bit_cast(unsigned int, f);
    unsigned int r = x + 0x7FFFu + ((x >> 16) & 1u);   // round-to-nearest-even
    return (unsigned short)(r >> 16);
}
__device__ __forceinline__ float bf2f(unsigned short s) {
    unsigned int x = ((unsigned int)s) << 16;
    return __builtin_bit_cast(float, x);
}
__device__ __forceinline__ v8f v8zero() {
    v8f z = {0.f, 0.f, 0.f, 0.f, 0.f, 0.f, 0.f, 0.f};
    return z;
}

// 16-byte global -> LDS copy: async on gfx1250, sync fallback elsewhere.
__device__ __forceinline__ void async_copy16(const unsigned short* g, unsigned short* l) {
#if TSF_ASYNC
    __builtin_amdgcn_global_load_async_to_lds_b128(
        (ASG v4i_t*)(g), (ASL v4i_t*)(l), 0, 0);
#else
    *reinterpret_cast<uint4*>(l) = *reinterpret_cast<const uint4*>(g);
#endif
}
__device__ __forceinline__ void async_wait0() {
#if TSF_ASYNC
    __builtin_amdgcn_s_wait_asynccnt(0);
#endif
}

#define TSF_T     16384
#define TSF_D     522
#define TSF_DP    544      // D padded to multiple of 32 (WMMA K granularity)
#define TSF_INNER 512
#define TSF_QKVN  1536

// ---------------------------------------------------------------------------
// GEMM: C[M,N] = A[M,Kp](bf16,row-major,lda) x W(bf16, pre-transposed N-major:
// Bt[Np][Kp], Np = ceil(N/128)*128, zero padded). Block tile 128x128, 8 waves
// (2x4), wave tile 64x32 -> 8 WMMAs / k-step. Double-buffered async staging.
// Epilogue: + bias[n], + Res[m*N+n]; writes fp32 Cf and/or bf16 Cb.
// ---------------------------------------------------------------------------
__global__ __launch_bounds__(256) void tsf_gemm_bf16(
    const unsigned short* __restrict__ A, int lda,
    const unsigned short* __restrict__ Bt, int N, int Kp,
    const float* __restrict__ bias, const float* __restrict__ Res,
    float* __restrict__ Cf, unsigned short* __restrict__ Cb)
{
    __shared__ unsigned short As[2][128][40];   // [buf][m][k], row 80B (16B-aligned)
    __shared__ unsigned short Bs[2][128][40];   // [buf][n][k]

    const int tid  = threadIdx.x;
    const int lane = tid & 31;
    const int wave = tid >> 5;
    const int wm   = wave >> 2;       // 0..1
    const int wn   = wave & 3;        // 0..3
    const int hi   = lane >> 4;       // half-wave
    const int ln   = lane & 15;
    const long m0  = (long)blockIdx.x * 128;
    const long n0  = (long)blockIdx.y * 128;

    const int r  = tid >> 1;          // 0..127 (tile row for A, tile col for B)
    const int c  = (tid & 1) << 4;    // 0 or 16

    v8f acc[4][2];
#pragma unroll
    for (int i = 0; i < 4; ++i)
#pragma unroll
        for (int j = 0; j < 2; ++j) acc[i][j] = v8zero();

    const int nsteps = Kp >> 5;

    // prologue: stage tile 0 into buffer 0
    {
        const unsigned short* ga = A  + (m0 + r) * lda + c;
        const unsigned short* gb = Bt + (n0 + r) * Kp  + c;
        async_copy16(ga,     &As[0][r][c]);
        async_copy16(ga + 8, &As[0][r][c + 8]);
        async_copy16(gb,     &Bs[0][r][c]);
        async_copy16(gb + 8, &Bs[0][r][c + 8]);
    }

    for (int s = 0; s < nsteps; ++s) {
        const int buf = s & 1;
        async_wait0();        // this wave's staged tile has landed in LDS
        __syncthreads();      // ... and every other wave's too

        if (s + 1 < nsteps) {   // stage next tile into the other buffer
            const int kk = (s + 1) << 5;
            const unsigned short* ga = A  + (m0 + r) * lda + kk + c;
            const unsigned short* gb = Bt + (n0 + r) * Kp  + kk + c;
            async_copy16(ga,     &As[buf ^ 1][r][c]);
            async_copy16(ga + 8, &As[buf ^ 1][r][c + 8]);
            async_copy16(gb,     &Bs[buf ^ 1][r][c]);
            async_copy16(gb + 8, &Bs[buf ^ 1][r][c + 8]);
        }

        FragB16 aF[4], bF[2];
#pragma unroll
        for (int ms = 0; ms < 4; ++ms) {
            int row = wm * 64 + ms * 16 + ln;
#pragma unroll
            for (int j = 0; j < 8; ++j) {
                int k = 2 * j + ((j >= 4) ? 8 : 0) + 8 * hi;   // A-frag K mapping
                aF[ms].w[j] = *reinterpret_cast<const unsigned int*>(&As[buf][row][k]);
            }
        }
#pragma unroll
        for (int ns = 0; ns < 2; ++ns) {
            int col = wn * 32 + ns * 16 + ln;
#pragma unroll
            for (int j = 0; j < 8; ++j) {
                int k = 2 * j + ((j >= 4) ? 8 : 0) + 8 * hi;   // B-frag K mapping
                bF[ns].w[j] = *reinterpret_cast<const unsigned int*>(&Bs[buf][col][k]);
            }
        }
#pragma unroll
        for (int ms = 0; ms < 4; ++ms)
#pragma unroll
            for (int ns = 0; ns < 2; ++ns)
                acc[ms][ns] = __builtin_amdgcn_wmma_f32_16x16x32_bf16(
                    false, aF[ms].v, false, bF[ns].v, (short)0, acc[ms][ns], false, false);
    }

#pragma unroll
    for (int ms = 0; ms < 4; ++ms) {
#pragma unroll
        for (int ns = 0; ns < 2; ++ns) {
            int n = (int)n0 + wn * 32 + ns * 16 + ln;
            if (n < N) {
#pragma unroll
                for (int rr = 0; rr < 8; ++rr) {
                    long m = m0 + wm * 64 + ms * 16 + rr + 8 * hi;
                    float v = acc[ms][ns][rr];
                    if (bias) v += bias[n];
                    if (Res)  v += Res[m * N + n];
                    if (Cf)   Cf[m * N + n] = v;
                    if (Cb)   Cb[m * N + n] = f2bf(v);
                }
            }
        }
    }
}

// ---------------------------------------------------------------------------
// Time attention: 8192 sequences of length 16 (batch = head*1024 + ni).
// One wave per sequence: 2 WMMA (QK^T) + softmax + 4 WMMA (P.V).
// Q/K/V laid out (HEADS, T, 64) bf16. O written (T, 512) bf16.
// ---------------------------------------------------------------------------
__global__ __launch_bounds__(256) void tsf_attn_time(
    const unsigned short* __restrict__ Q, const unsigned short* __restrict__ K,
    const unsigned short* __restrict__ V, unsigned short* __restrict__ O)
{
    __shared__ float Plds[8][16][16];
    const int tid = threadIdx.x, lane = tid & 31, w = tid >> 5;
    const int hi = lane >> 4, ln = lane & 15;
    const int s  = blockIdx.x * 8 + w;
    const int h  = s >> 10, ni = s & 1023;
    const long hTd = (long)h * TSF_T * 64;

    FragB16 aQ[2];
    {
        long base = hTd + ((long)ln * 1024 + ni) * 64;   // query row fi = ln
#pragma unroll
        for (int kb = 0; kb < 2; ++kb)
#pragma unroll
            for (int j = 0; j < 8; ++j) {
                int k = 2 * j + ((j >= 4) ? 8 : 0) + 8 * hi + 32 * kb;
                aQ[kb].w[j] = *reinterpret_cast<const unsigned int*>(Q + base + k);
            }
    }
    v8f sim = v8zero();
#pragma unroll
    for (int kb = 0; kb < 2; ++kb) {
        FragB16 bK;
        long base = hTd + ((long)ln * 1024 + ni) * 64;    // key row fj = ln
#pragma unroll
        for (int j = 0; j < 8; ++j) {
            int k = 2 * j + ((j >= 4) ? 8 : 0) + 8 * hi + 32 * kb;
            bK.w[j] = *reinterpret_cast<const unsigned int*>(K + base + k);
        }
        sim = __builtin_amdgcn_wmma_f32_16x16x32_bf16(
            false, aQ[kb].v, false, bK.v, (short)0, sim, false, false);
    }
    float p[8];
#pragma unroll
    for (int r = 0; r < 8; ++r) {
        float v = sim[r];
        float m = v;
        for (int o = 1; o < 16; o <<= 1) m = fmaxf(m, __shfl_xor(m, o, 32));
        float e = __expf(v - m);
        float su = e;
        for (int o = 1; o < 16; o <<= 1) su += __shfl_xor(su, o, 32);
        p[r] = e / su;
    }
#pragma unroll
    for (int r = 0; r < 8; ++r) Plds[w][r + 8 * hi][ln] = p[r];
    FragB16 aP;
#pragma unroll
    for (int e = 0; e < 16; ++e) {
        int k = e + 8 * (e >> 3) + 8 * hi;
        float pv = (k < 16) ? Plds[w][ln][k] : 0.0f;
        aP.u[e] = f2bf(pv);
    }
#pragma unroll
    for (int nt = 0; nt < 4; ++nt) {
        FragB16 bV;
#pragma unroll
        for (int e = 0; e < 16; ++e) {
            int k = e + 8 * (e >> 3) + 8 * hi;
            bV.u[e] = (k < 16) ? V[hTd + ((long)k * 1024 + ni) * 64 + nt * 16 + ln]
                               : (unsigned short)0;
        }
        v8f o = v8zero();
        o = __builtin_amdgcn_wmma_f32_16x16x32_bf16(
            false, aP.v, false, bV.v, (short)0, o, false, false);
#pragma unroll
        for (int r = 0; r < 8; ++r) {
            long t = (long)(r + 8 * hi) * 1024 + ni;
            O[t * TSF_INNER + h * 64 + nt * 16 + ln] = f2bf(o[r]);
        }
    }
}

// ---------------------------------------------------------------------------
// Space attention: 128 sequences of length 1024 (seq = head*16 + fi).
// Flash-style online softmax; one wave owns 16 query rows. K/V 16x64 tiles
// are shared by all 8 waves -> staged cooperatively via async-to-LDS.
// Per 16-key tile: 2 WMMA (QK^T) + 4 WMMA (P.V).
// ---------------------------------------------------------------------------
__global__ __launch_bounds__(256) void tsf_attn_space(
    const unsigned short* __restrict__ Q, const unsigned short* __restrict__ K,
    const unsigned short* __restrict__ V, unsigned short* __restrict__ O)
{
    __shared__ float Plds[8][16][16];
    __shared__ unsigned short Ks[16][80];   // row 160B (16B-aligned)
    __shared__ unsigned short Vs[16][80];
    const int tid = threadIdx.x, lane = tid & 31, w = tid >> 5;
    const int hi = lane >> 4, ln = lane & 15;
    const int blk = blockIdx.x;
    const int seq = blk >> 3;          // 0..127
    const int qb  = blk & 7;
    const int h   = seq >> 4, fi = seq & 15;
    const int nbase = qb * 128 + w * 16;
    const long seqBase = (long)h * TSF_T * 64 + (long)fi * 1024 * 64;

    FragB16 aQ[2];
    {
        long base = seqBase + (long)(nbase + ln) * 64;
#pragma unroll
        for (int kb = 0; kb < 2; ++kb)
#pragma unroll
            for (int j = 0; j < 8; ++j) {
                int k = 2 * j + ((j >= 4) ? 8 : 0) + 8 * hi + 32 * kb;
                aQ[kb].w[j] = *reinterpret_cast<const unsigned int*>(Q + base + k);
            }
    }

    // cooperative K/V tile staging: waves 0-3 load K, waves 4-7 load V
    const int qq  = tid & 127;
    const int trow = qq >> 3;            // 0..15
    const int tseg = (qq & 7) * 8;       // 0,8,...,56 (bf16 elements)
    const unsigned short* gsrcB = (tid < 128) ? K : V;
    unsigned short* ldst = (tid < 128) ? &Ks[trow][tseg] : &Vs[trow][tseg];

    float m_i[8], l_i[8];
    v8f accO[4];
#pragma unroll
    for (int r = 0; r < 8; ++r) { m_i[r] = -1e30f; l_i[r] = 0.0f; }
#pragma unroll
    for (int nt = 0; nt < 4; ++nt) accO[nt] = v8zero();

    for (int jt = 0; jt < 64; ++jt) {
        const int j0 = jt * 16;
        __syncthreads();   // everyone done reading previous K/V tile
        async_copy16(gsrcB + seqBase + (long)(j0 + trow) * 64 + tseg, ldst);
        async_wait0();
        __syncthreads();   // tile visible to all waves

        v8f sim = v8zero();
#pragma unroll
        for (int kb = 0; kb < 2; ++kb) {
            FragB16 bK;
#pragma unroll
            for (int j = 0; j < 8; ++j) {
                int k = 2 * j + ((j >= 4) ? 8 : 0) + 8 * hi + 32 * kb;
                bK.w[j] = *reinterpret_cast<const unsigned int*>(&Ks[ln][k]);
            }
            sim = __builtin_amdgcn_wmma_f32_16x16x32_bf16(
                false, aQ[kb].v, false, bK.v, (short)0, sim, false, false);
        }
#pragma unroll
        for (int r = 0; r < 8; ++r) {
            float v = sim[r];
            float mx = v;
            for (int o = 1; o < 16; o <<= 1) mx = fmaxf(mx, __shfl_xor(mx, o, 32));
            float mnew = fmaxf(m_i[r], mx);
            float p  = __expf(v - mnew);
            float rs = p;
            for (int o = 1; o < 16; o <<= 1) rs += __shfl_xor(rs, o, 32);
            float scl = __expf(m_i[r] - mnew);
            l_i[r] = l_i[r] * scl + rs;
            m_i[r] = mnew;
#pragma unroll
            for (int nt = 0; nt < 4; ++nt) accO[nt][r] *= scl;
            Plds[w][r + 8 * hi][ln] = p;
        }
        FragB16 aP;
#pragma unroll
        for (int e = 0; e < 16; ++e) {
            int k = e + 8 * (e >> 3) + 8 * hi;
            float pv = (k < 16) ? Plds[w][ln][k] : 0.0f;
            aP.u[e] = f2bf(pv);
        }
#pragma unroll
        for (int nt = 0; nt < 4; ++nt) {
            FragB16 bV;
#pragma unroll
            for (int e = 0; e < 16; ++e) {
                int k = e + 8 * (e >> 3) + 8 * hi;
                bV.u[e] = (k < 16) ? Vs[k][nt * 16 + ln] : (unsigned short)0;
            }
            accO[nt] = __builtin_amdgcn_wmma_f32_16x16x32_bf16(
                false, aP.v, false, bV.v, (short)0, accO[nt], false, false);
        }
    }

#pragma unroll
    for (int nt = 0; nt < 4; ++nt)
#pragma unroll
        for (int r = 0; r < 8; ++r) {
            int ni = nbase + r + 8 * hi;
            long t = (long)fi * 1024 + ni;
            O[t * TSF_INNER + h * 64 + nt * 16 + ln] = f2bf(accO[nt][r] / l_i[r]);
        }
}

// ---------------------------------------------------------------------------
// Elementwise / prep kernels
// ---------------------------------------------------------------------------
__global__ void tsf_init_tables(float* fsin, float* fcos, float* isin, float* icos)
{
    int idx = blockIdx.x * blockDim.x + threadIdx.x;   // 1024*64
    int n = idx >> 6, d = idx & 63;
    int j = d >> 1;
    const float PI = 3.14159265358979323846f;
    const float L25 = 2.3219280948873623f / 15.0f;     // log2(5)/15
    float v;
    if (j < 16) {
        int a = n >> 5;
        v = (a * (2.0f / 31.0f) - 1.0f) * exp2f(j * L25) * PI;
    } else {
        int a = n & 31;
        v = (a * (2.0f / 31.0f) - 1.0f) * exp2f((j - 16) * L25) * PI;
    }
    isin[idx] = sinf(v);
    icos[idx] = cosf(v);
    if (n < 16) {
        int i = d & 31;
        float inv = exp2f(-((float)i / 32.0f) * 13.287712379549449f); // 10000^(-i/32)
        float fv = (float)n * inv;
        fsin[n * 64 + d] = sinf(fv);
        fcos[n * 64 + d] = cosf(fv);
    }
}

__global__ void tsf_patch_embed(const float* __restrict__ video, const int* __restrict__ cond,
                                const float* __restrict__ pw, const float* __restrict__ pb,
                                const float* __restrict__ label, float* __restrict__ X)
{
    long idx = (long)blockIdx.x * blockDim.x + threadIdx.x;
    if (idx >= (long)TSF_T * TSF_D) return;
    int t = (int)(idx / TSF_D), j = (int)(idx % TSF_D);
    if (j < 512) {
        int hh = t >> 9, rem = t & 511, ff = rem >> 5, ww = rem & 31;
        float s = pb[j];
#pragma unroll
        for (int c = 0; c < 8; ++c)
            s += video[(((long)ff * 8 + c) * 32 + hh) * 32 + ww] * pw[c * 512 + j];
        X[idx] = s;
    } else {
        X[idx] = label[(long)cond[0] * TSF_T + t];   // broadcast over lc=10 cols
    }
}

__global__ __launch_bounds__(256) void tsf_layernorm(
    const float* __restrict__ X, const float* __restrict__ g, const float* __restrict__ b,
    unsigned short* __restrict__ Xn)
{
    __shared__ float red[256];
    __shared__ float stats[2];
    const int t = blockIdx.x, tid = threadIdx.x;
    const float* xr = X + (long)t * TSF_D;
    float s = 0.f;
    for (int j = tid; j < TSF_D; j += 256) s += xr[j];
    red[tid] = s; __syncthreads();
    for (int o = 128; o > 0; o >>= 1) { if (tid < o) red[tid] += red[tid + o]; __syncthreads(); }
    if (tid == 0) stats[0] = red[0] / (float)TSF_D;
    __syncthreads();
    float mean = stats[0];
    float vs = 0.f;
    for (int j = tid; j < TSF_D; j += 256) { float d = xr[j] - mean; vs += d * d; }
    __syncthreads();
    red[tid] = vs; __syncthreads();
    for (int o = 128; o > 0; o >>= 1) { if (tid < o) red[tid] += red[tid + o]; __syncthreads(); }
    if (tid == 0) stats[1] = rsqrtf(red[0] / (float)TSF_D + 1e-5f);
    __syncthreads();
    float inv = stats[1];
    unsigned short* xo = Xn + (long)t * TSF_DP;
    for (int j = tid; j < TSF_DP; j += 256) {
        float v = (j < TSF_D) ? (xr[j] - mean) * inv * g[j] + b[j] : 0.0f;
        xo[j] = f2bf(v);
    }
}

// QKV (T,1536) bf16 -> scaled+rotary Q/K and V in (HEADS,T,64) bf16 layout.
__global__ void tsf_qkv_prep(const unsigned short* __restrict__ QKV,
                             unsigned short* __restrict__ Qo, unsigned short* __restrict__ Ko,
                             unsigned short* __restrict__ Vo,
                             const float* __restrict__ sinT, const float* __restrict__ cosT,
                             int timeMode)
{
    long idx = (long)blockIdx.x * blockDim.x + threadIdx.x;   // HEADS*T*32
    int p = (int)(idx & 31);
    int t = (int)((idx >> 5) & (TSF_T - 1));
    int h = (int)(idx >> 19);
    int d0 = 2 * p, d1 = d0 + 1;
    long base = (long)t * TSF_QKVN + h * 64;
    float q0 = bf2f(QKV[base + d0]) * 0.125f;        // DH^-0.5 = 1/8
    float q1 = bf2f(QKV[base + d1]) * 0.125f;
    float k0 = bf2f(QKV[base + 512 + d0]);
    float k1 = bf2f(QKV[base + 512 + d1]);
    int pos = timeMode ? (t >> 10) : (t & 1023);
    float c0 = cosT[pos * 64 + d0], c1 = cosT[pos * 64 + d1];
    float s0 = sinT[pos * 64 + d0], s1 = sinT[pos * 64 + d1];
    float qr0 = q0 * c0 - q1 * s0, qr1 = q1 * c1 + q0 * s1;
    float kr0 = k0 * c0 - k1 * s0, kr1 = k1 * c1 + k0 * s1;
    long ob = (long)h * TSF_T * 64 + (long)t * 64 + d0;
    Qo[ob] = f2bf(qr0); Qo[ob + 1] = f2bf(qr1);
    Ko[ob] = f2bf(kr0); Ko[ob + 1] = f2bf(kr1);
    Vo[ob] = QKV[base + 1024 + d0]; Vo[ob + 1] = QKV[base + 1024 + d1];
}

__global__ void tsf_geglu(const float* __restrict__ H, unsigned short* __restrict__ G)
{
    long idx = (long)blockIdx.x * blockDim.x + threadIdx.x;   // T*256
    int t = (int)(idx >> 8), j = (int)(idx & 255);
    float a = H[(long)t * 512 + j];
    float x = H[(long)t * 512 + 256 + j];
    float gl = 0.5f * x * (1.0f + erff(x * 0.7071067811865476f));
    G[idx] = f2bf(a * gl);
}

__global__ void tsf_cvt_x(const float* __restrict__ X, unsigned short* __restrict__ Xb)
{
    long idx = (long)blockIdx.x * blockDim.x + threadIdx.x;   // T*544
    int t = (int)(idx / TSF_DP), j = (int)(idx % TSF_DP);
    Xb[idx] = (j < TSF_D) ? f2bf(X[(long)t * TSF_D + j]) : (unsigned short)0;
}

// Weight convert: W[Ksrc x N] fp32 -> Wb[Np x Kp] bf16, TRANSPOSED (N-major),
// zero-padded in both K (to Kp) and N (to Np).
__global__ void tsf_wconv_t(const float* __restrict__ W, unsigned short* __restrict__ Wb,
                            int Ksrc, int Kp, int N, int Np)
{
    long idx = (long)blockIdx.x * blockDim.x + threadIdx.x;
    if (idx >= (long)Np * Kp) return;
    int n = (int)(idx / Kp), k = (int)(idx % Kp);
    Wb[idx] = (k < Ksrc && n < N) ? f2bf(W[(long)k * N + n]) : (unsigned short)0;
}

// ---------------------------------------------------------------------------
// Orchestration
// ---------------------------------------------------------------------------
extern "C" void kernel_launch(void* const* d_in, const int* in_sizes, int n_in,
                              void* d_out, int out_size, void* d_ws, size_t ws_size,
                              hipStream_t stream)
{
    (void)in_sizes; (void)n_in; (void)out_size; (void)ws_size;
    const float* video    = (const float*)d_in[0];
    const int*   cond     = (const int*)d_in[1];
    const float* patch_w  = (const float*)d_in[3];
    const float* patch_b  = (const float*)d_in[4];
    const float* label    = (const float*)d_in[5];
    const float* ln_t_g   = (const float*)d_in[6];
    const float* ln_t_b   = (const float*)d_in[7];
    const float* qkv_t    = (const float*)d_in[8];
    const float* out_t_w  = (const float*)d_in[9];
    const float* out_t_b  = (const float*)d_in[10];
    const float* ln_s_g   = (const float*)d_in[11];
    const float* ln_s_b   = (const float*)d_in[12];
    const float* qkv_s    = (const float*)d_in[13];
    const float* out_s_w  = (const float*)d_in[14];
    const float* out_s_b  = (const float*)d_in[15];
    const float* ln_f_g   = (const float*)d_in[16];
    const float* ln_f_b   = (const float*)d_in[17];
    const float* ff_w1    = (const float*)d_in[18];
    const float* ff_b1    = (const float*)d_in[19];
    const float* ff_w2    = (const float*)d_in[20];
    const float* ff_b2    = (const float*)d_in[21];
    const float* fc_w     = (const float*)d_in[22];
    const float* fc_b     = (const float*)d_in[23];

    char* ws = (char*)d_ws;
    const long T = TSF_T;
    size_t off_X   = 0;                                     // fp32 T x 522
    size_t off_A   = off_X   + (size_t)T * TSF_D * 4;       // bf16 T x 544 (Xn / O / G / Xb)
    size_t off_QKV = off_A   + (size_t)T * TSF_DP * 2;      // bf16 T x 1536 (also fp32 T x 512 FFN hidden)
    size_t off_Q   = off_QKV + (size_t)T * TSF_QKVN * 2;    // bf16 HEADS x T x 64
    size_t off_K   = off_Q   + (size_t)T * TSF_INNER * 2;
    size_t off_V   = off_K   + (size_t)T * TSF_INNER * 2;
    size_t off_W   = off_V   + (size_t)T * TSF_INNER * 2;   // bf16 weight arena (<= 1536*544)
    size_t off_FS  = off_W   + (size_t)TSF_QKVN * TSF_DP * 2;
    size_t off_FC  = off_FS  + 16 * 64 * 4;
    size_t off_IS  = off_FC  + 16 * 64 * 4;
    size_t off_IC  = off_IS  + 1024 * 64 * 4;

    float*          X    = (float*)(ws + off_X);
    unsigned short* Abuf = (unsigned short*)(ws + off_A);
    unsigned short* QKVb = (unsigned short*)(ws + off_QKV);
    float*          Hf   = (float*)(ws + off_QKV);
    unsigned short* Qb   = (unsigned short*)(ws + off_Q);
    unsigned short* Kb   = (unsigned short*)(ws + off_K);
    unsigned short* Vb   = (unsigned short*)(ws + off_V);
    unsigned short* Wb   = (unsigned short*)(ws + off_W);
    float* fsin = (float*)(ws + off_FS);
    float* fcos = (float*)(ws + off_FC);
    float* isin = (float*)(ws + off_IS);
    float* icos = (float*)(ws + off_IC);

    auto gemm = [&](const unsigned short* A, int lda, int N, int Kp,
                    const float* bias, const float* Res,
                    float* Cf, unsigned short* Cb) {
        dim3 g(128, (unsigned)((N + 127) / 128));
        tsf_gemm_bf16<<<g, 256, 0, stream>>>(A, lda, Wb, N, Kp, bias, Res, Cf, Cb);
    };
    auto wconv = [&](const float* W, int Ksrc, int Kp, int N) {
        int Np = ((N + 127) / 128) * 128;
        long tot = (long)Np * Kp;
        tsf_wconv_t<<<(int)((tot + 255) / 256), 256, 0, stream>>>(W, Wb, Ksrc, Kp, N, Np);
    };

    tsf_init_tables<<<256, 256, 0, stream>>>(fsin, fcos, isin, icos);
    {
        long tot = T * TSF_D;
        tsf_patch_embed<<<(int)((tot + 255) / 256), 256, 0, stream>>>(
            video, cond, patch_w, patch_b, label, X);
    }

    for (int L = 0; L < 2; ++L) {
        // ---- time attention block ----
        tsf_layernorm<<<(int)T, 256, 0, stream>>>(X, ln_t_g + L * TSF_D, ln_t_b + L * TSF_D, Abuf);
        wconv(qkv_t + (long)L * TSF_D * TSF_QKVN, TSF_D, TSF_DP, TSF_QKVN);
        gemm(Abuf, TSF_DP, TSF_QKVN, TSF_DP, nullptr, nullptr, nullptr, QKVb);
        tsf_qkv_prep<<<16384, 256, 0, stream>>>(QKVb, Qb, Kb, Vb, fsin, fcos, 1);
        tsf_attn_time<<<1024, 256, 0, stream>>>(Qb, Kb, Vb, Abuf);
        wconv(out_t_w + (long)L * TSF_INNER * TSF_D, TSF_INNER, TSF_INNER, TSF_D);
        gemm(Abuf, TSF_INNER, TSF_D, TSF_INNER, out_t_b + L * TSF_D, X, X, nullptr);

        // ---- space attention block ----
        tsf_layernorm<<<(int)T, 256, 0, stream>>>(X, ln_s_g + L * TSF_D, ln_s_b + L * TSF_D, Abuf);
        wconv(qkv_s + (long)L * TSF_D * TSF_QKVN, TSF_D, TSF_DP, TSF_QKVN);
        gemm(Abuf, TSF_DP, TSF_QKVN, TSF_DP, nullptr, nullptr, nullptr, QKVb);
        tsf_qkv_prep<<<16384, 256, 0, stream>>>(QKVb, Qb, Kb, Vb, isin, icos, 0);
        tsf_attn_space<<<1024, 256, 0, stream>>>(Qb, Kb, Vb, Abuf);
        wconv(out_s_w + (long)L * TSF_INNER * TSF_D, TSF_INNER, TSF_INNER, TSF_D);
        gemm(Abuf, TSF_INNER, TSF_D, TSF_INNER, out_s_b + L * TSF_D, X, X, nullptr);

        // ---- GEGLU FFN ----
        tsf_layernorm<<<(int)T, 256, 0, stream>>>(X, ln_f_g + L * TSF_D, ln_f_b + L * TSF_D, Abuf);
        wconv(ff_w1 + (long)L * TSF_D * 512, TSF_D, TSF_DP, 512);
        gemm(Abuf, TSF_DP, 512, TSF_DP, ff_b1 + L * 512, nullptr, Hf, nullptr);
        tsf_geglu<<<16384, 256, 0, stream>>>(Hf, Abuf);
        wconv(ff_w2 + (long)L * 256 * TSF_D, 256, 256, TSF_D);
        gemm(Abuf, 256, TSF_D, 256, ff_b2 + L * TSF_D, X, X, nullptr);
    }

    // ---- final projection ----
    {
        long tot = T * TSF_DP;
        tsf_cvt_x<<<(int)((tot + 255) / 256), 256, 0, stream>>>(X, Abuf);
    }
    wconv(fc_w, TSF_D, TSF_DP, 64);
    gemm(Abuf, TSF_DP, 64, TSF_DP, fc_b, nullptr, (float*)d_out, nullptr);
}
